// AttnWrapper_18330920419364
// MI455X (gfx1250) — matrix operand order repaired and checked by reading, not compiled, not run on hardware
//
#include <hip/hip_runtime.h>
#include <hip/hip_bf16.h>

// ---------------------------------------------------------------------------
// Problem constants (from reference: B=4, L=2048, D=512, H=8, W=32, OVERLAP=32)
// ---------------------------------------------------------------------------
constexpr int Bc      = 4;
constexpr int Lc      = 2048;
constexpr int Dc      = 512;
constexpr int Hc      = 8;
constexpr int HDc     = 64;            // head dim
constexpr int OVL     = 32;            // OVERLAP
constexpr int LP      = Lc + 2 * OVL;  // 2112 padded rows per batch
constexpr int MP      = Bc * LP;       // 8448 padded rows total
constexpr int MQ      = Bc * Lc;       // 8192 main rows total
constexpr int NQKV    = 3 * Dc;        // 1536

// ---------------------------------------------------------------------------
// Workspace layout (bytes, 256-aligned)
// ---------------------------------------------------------------------------
constexpr size_t XBF_OFF   = 0;
constexpr size_t XBF_BYTES = (size_t)MP * Dc * 2;            // bf16 padded input
constexpr size_t WQKV_OFF  = XBF_OFF + XBF_BYTES;
constexpr size_t WQKV_BYTES= (size_t)NQKV * Dc * 2;          // bf16 in_proj_w
constexpr size_t WOUT_OFF  = WQKV_OFF + WQKV_BYTES;
constexpr size_t WOUT_BYTES= (size_t)Dc * Dc * 2;            // bf16 out_proj_w
constexpr size_t QKV_OFF   = WOUT_OFF + WOUT_BYTES;
constexpr size_t QKV_BYTES = (size_t)MP * NQKV * 4;          // fp32 qkv
constexpr size_t OBF_OFF   = QKV_OFF + QKV_BYTES;
constexpr size_t OBF_BYTES = (size_t)MQ * Dc * 2;            // bf16 attn output

// ---------------------------------------------------------------------------
// WMMA types (CDNA5: wave32, v_wmma_f32_16x16x32_bf16)
// ---------------------------------------------------------------------------
typedef __attribute__((ext_vector_type(16))) __bf16 v16bf;
typedef __attribute__((ext_vector_type(8)))  float  v8f;

union FragBF {
  v16bf v;
  uint4 q[2];
};

__device__ __forceinline__ unsigned short f32_to_bf16(float f) {
  unsigned int u = __float_as_uint(f);
  unsigned int r = (u + 0x7FFFu + ((u >> 16) & 1u)) >> 16;   // RNE
  return (unsigned short)r;
}

// ---------------------------------------------------------------------------
// Pack kernels
// ---------------------------------------------------------------------------
__global__ void __launch_bounds__(256)
pack_x_kernel(const float* __restrict__ beg, const float* __restrict__ mainp,
              const float* __restrict__ endp, unsigned short* __restrict__ xbf) {
  int i = blockIdx.x * 256 + threadIdx.x;
  if (i >= MP * Dc) return;
  int b   = i / (LP * Dc);
  int rem = i - b * (LP * Dc);
  int r   = rem >> 9;           // Dc = 512
  int d   = rem & (Dc - 1);
  float v;
  if (r < OVL)            v = beg  [(size_t)(b * OVL + r) * Dc + d];
  else if (r < OVL + Lc)  v = mainp[(size_t)(b * Lc + (r - OVL)) * Dc + d];
  else                    v = endp [(size_t)(b * OVL + (r - OVL - Lc)) * Dc + d];
  xbf[i] = f32_to_bf16(v);
}

__global__ void __launch_bounds__(256)
pack_w_kernel(const float* __restrict__ src, unsigned short* __restrict__ dst, int n) {
  int i = blockIdx.x * 256 + threadIdx.x;
  if (i < n) dst[i] = f32_to_bf16(src[i]);
}

// fp32 residual copy: out[row, 0:512] = main[row, :]
__global__ void __launch_bounds__(256)
copy_main_kernel(const float* __restrict__ mainp, float* __restrict__ out) {
  int i = blockIdx.x * 256 + threadIdx.x;        // float4 index
  if (i >= MQ * Dc / 4) return;
  int row = i >> 7;                              // 128 float4 per row
  int c4  = i & 127;
  float4 v = ((const float4*)mainp)[i];
  ((float4*)(out + (size_t)row * (2 * Dc)))[c4] = v;
}

// ---------------------------------------------------------------------------
// Fragment helpers (layouts per CDNA5 ISA 7.12.2, wave32)
// ---------------------------------------------------------------------------
__device__ __forceinline__ void load_frags(FragBF (&a)[2], FragBF (&b)[4],
                                           const unsigned short* arow0,
                                           const unsigned short* arow1,
                                           const unsigned short* const (&wr)[4],
                                           int k0) {
  a[0].q[0] = *(const uint4*)(arow0 + k0);
  a[0].q[1] = *(const uint4*)(arow0 + k0 + 16);
  a[1].q[0] = *(const uint4*)(arow1 + k0);
  a[1].q[1] = *(const uint4*)(arow1 + k0 + 16);
#pragma unroll
  for (int j = 0; j < 4; ++j) {
    b[j].q[0] = *(const uint4*)(wr[j] + k0);
    b[j].q[1] = *(const uint4*)(wr[j] + k0 + 16);
  }
}

__device__ __forceinline__ void mma_tile(v8f (&acc)[2][4],
                                         const FragBF (&a)[2],
                                         const FragBF (&b)[4]) {
#pragma unroll
  for (int j = 0; j < 4; ++j) {
#pragma unroll
    for (int i = 0; i < 2; ++i)
      acc[i][j] = __builtin_amdgcn_wmma_f32_16x16x32_bf16(
          /*neg_a=*/false, a[i].v, /*neg_b=*/false, b[j].v,
          /*c_mod=*/(short)0, acc[i][j], /*reuse_a=*/false, /*reuse_b=*/false);
  }
}

// ---------------------------------------------------------------------------
// bf16 GEMM: C[M,N] (fp32, stride ldc) = A[M,K](bf16) @ Wt[N,K]^T(bf16) + bias
// One wave computes a 32x64 tile (2 A-strips x 4 B-subtiles, 8 accumulators).
// Register double-buffered pipeline: fragments for k-step s+1 are in flight
// while the 8 WMMAs for k-step s issue, so waits cover only the prior batch.
// ---------------------------------------------------------------------------
__global__ void __launch_bounds__(256)
gemm_bf16_wmma(const unsigned short* __restrict__ A, int lda,
               const unsigned short* __restrict__ Wt, int ldw,
               const float* __restrict__ bias,
               float* __restrict__ C, int ldc,
               int M, int K, int tiles_n) {
  const int lane = threadIdx.x & 31;
  const int wave = threadIdx.x >> 5;
  const int tile = blockIdx.x * 8 + wave;
  const int mt   = tile / tiles_n;
  const int nt   = tile - mt * tiles_n;
  const int m0   = mt * 32;
  const int n0   = nt * 64;
  if (m0 >= M) return;

  const int lm  = lane & 15;
  const int khA = (lane >> 4) * 8;     // A: lanes 0-15 -> K 0..7/16..23, 16-31 -> 8..15/24..31
  const int khB = (lane >> 4) * 8;     // B: lanes 0-15 -> K 0..7/16..23, 16-31 -> 8..15/24..31

  const unsigned short* arow0 = A + (size_t)(m0 + lm) * lda + khA;
  const unsigned short* arow1 = arow0 + (size_t)16 * lda;
  const unsigned short* wr[4];
#pragma unroll
  for (int j = 0; j < 4; ++j)
    wr[j] = Wt + (size_t)(n0 + j * 16 + lm) * ldw + khB;

  v8f acc[2][4];
#pragma unroll
  for (int i = 0; i < 2; ++i)
#pragma unroll
    for (int j = 0; j < 4; ++j)
      acc[i][j] = (v8f){0.f, 0.f, 0.f, 0.f, 0.f, 0.f, 0.f, 0.f};

  FragBF a0[2], b0[4], a1[2], b1[4];
  const int nsteps = K >> 5;                       // K / 32
  load_frags(a0, b0, arow0, arow1, wr, 0);

  for (int ks = 0; ks < nsteps; ks += 2) {
    if (ks + 1 < nsteps)
      load_frags(a1, b1, arow0, arow1, wr, (ks + 1) << 5);
    mma_tile(acc, a0, b0);
    if (ks + 2 < nsteps)
      load_frags(a0, b0, arow0, arow1, wr, (ks + 2) << 5);
    if (ks + 1 < nsteps)
      mma_tile(acc, a1, b1);
  }

  // C/D layout: VGPR r -> M = base + r (lanes 0-15) / base + 8 + r (lanes 16-31)
#pragma unroll
  for (int i = 0; i < 2; ++i) {
    const int mrow = m0 + i * 16 + (lane >> 4) * 8;
#pragma unroll
    for (int j = 0; j < 4; ++j) {
      const int   n  = n0 + j * 16 + lm;
      const float bv = bias[n];
#pragma unroll
      for (int r = 0; r < 8; ++r)
        C[(size_t)(mrow + r) * ldc + n] = acc[i][j][r] + bv;
    }
  }
}

// ---------------------------------------------------------------------------
// Dilated local attention, one wave32 per (b, l, h).
// qkv rows: [B*2112, 1536] = [q | k | v]; query row index = b*2112 + 32 + l.
// Window offsets: w<16 -> 2w-32 (-32..-2), w>=16 -> 2w-30 (2..32); always
// in-bounds within the padded rows.
// ---------------------------------------------------------------------------
__global__ void __launch_bounds__(256)
attn_kernel(const float* __restrict__ qkv, unsigned short* __restrict__ obf) {
  const int lane = threadIdx.x & 31;
  const int wave = threadIdx.x >> 5;
  const int gid  = blockIdx.x * 8 + wave;     // (b*L + l)*H + h
  const int h = gid & (Hc - 1);
  const int l = (gid >> 3) & (Lc - 1);
  const int b = gid >> 14;                    // L*H = 16384 = 2^14

  const int  w    = lane;
  const int  off  = (w < 16) ? (2 * w - 32) : (2 * w - 30);
  const long rowq = (long)b * LP + OVL + l;

  const float* qrow = qkv + (size_t)rowq * NQKV + h * HDc;
  const float* krow = qkv + (size_t)(rowq + off) * NQKV + Dc + h * HDc;

  // scores: s_w = q . k_w / sqrt(64)
  float s = 0.f;
#pragma unroll
  for (int d = 0; d < HDc; d += 4) {
    float4 qa = *(const float4*)(qrow + d);
    float4 ka = *(const float4*)(krow + d);
    s = fmaf(qa.x, ka.x, s);
    s = fmaf(qa.y, ka.y, s);
    s = fmaf(qa.z, ka.z, s);
    s = fmaf(qa.w, ka.w, s);
  }
  s *= 0.125f;

  // wave32 softmax over the 32 window slots
  float mx = s;
#pragma unroll
  for (int o = 16; o > 0; o >>= 1) mx = fmaxf(mx, __shfl_xor(mx, o, 32));
  float e   = __expf(s - mx);
  float sum = e;
#pragma unroll
  for (int o = 16; o > 0; o >>= 1) sum += __shfl_xor(sum, o, 32);
  const float p = e / sum;

  // o[d] = sum_w p_w * v_w[d]; each lane owns dims (2*lane, 2*lane+1)
  float a0 = 0.f, a1 = 0.f;
#pragma unroll
  for (int w2 = 0; w2 < 32; ++w2) {
    const float pw   = __shfl(p, w2, 32);
    const int   offw = (w2 < 16) ? (2 * w2 - 32) : (2 * w2 - 30);
    const float* vrow = qkv + (size_t)(rowq + offw) * NQKV + 2 * Dc + h * HDc;
    float2 vv = *(const float2*)(vrow + 2 * lane);
    a0 = fmaf(pw, vv.x, a0);
    a1 = fmaf(pw, vv.y, a1);
  }

  unsigned int packed = (unsigned int)f32_to_bf16(a0) |
                        ((unsigned int)f32_to_bf16(a1) << 16);
  *(unsigned int*)(obf + ((size_t)(b * Lc + l) * Dc + h * HDc + 2 * lane)) = packed;
}

// ---------------------------------------------------------------------------
// Launch
// ---------------------------------------------------------------------------
extern "C" void kernel_launch(void* const* d_in, const int* in_sizes, int n_in,
                              void* d_out, int out_size, void* d_ws, size_t ws_size,
                              hipStream_t stream) {
  (void)in_sizes; (void)n_in; (void)out_size; (void)ws_size;

  const float* beg   = (const float*)d_in[0];
  const float* mainp = (const float*)d_in[1];
  const float* endp  = (const float*)d_in[2];
  const float* in_w  = (const float*)d_in[3];   // [1536, 512]
  const float* in_b  = (const float*)d_in[4];   // [1536]
  const float* out_w = (const float*)d_in[5];   // [512, 512]
  const float* out_b = (const float*)d_in[6];   // [512]
  float* out = (float*)d_out;                   // [8192, 1024]

  char* ws = (char*)d_ws;
  unsigned short* xbf   = (unsigned short*)(ws + XBF_OFF);
  unsigned short* wqkvb = (unsigned short*)(ws + WQKV_OFF);
  unsigned short* woutb = (unsigned short*)(ws + WOUT_OFF);
  float*          qkv   = (float*)(ws + QKV_OFF);
  unsigned short* obf   = (unsigned short*)(ws + OBF_OFF);

  // 1) pack bf16 operands + fp32 residual copy
  pack_x_kernel<<<(MP * Dc + 255) / 256, 256, 0, stream>>>(beg, mainp, endp, xbf);
  pack_w_kernel<<<(NQKV * Dc + 255) / 256, 256, 0, stream>>>(in_w, wqkvb, NQKV * Dc);
  pack_w_kernel<<<(Dc * Dc + 255) / 256, 256, 0, stream>>>(out_w, woutb, Dc * Dc);
  copy_main_kernel<<<(MQ * Dc / 4 + 255) / 256, 256, 0, stream>>>(mainp, out);

  // 2) fused QKV projection: [8448,512] @ [512,1536]^T + b -> qkv fp32
  {
    const int tiles_n = NQKV / 64;                 // 24
    const int waves   = (MP / 32) * tiles_n;       // 264 * 24 = 6336
    gemm_bf16_wmma<<<waves / 8, 256, 0, stream>>>(xbf, Dc, wqkvb, Dc, in_b,
                                                  qkv, NQKV, MP, Dc, tiles_n);
  }

  // 3) windowed attention (fp32 math, bf16 output for out-proj)
  attn_kernel<<<(Bc * Lc * Hc) / 8, 256, 0, stream>>>(qkv, obf);

  // 4) output projection straight into out[:, 512:1024]
  {
    const int tiles_n = Dc / 64;                   // 8
    const int waves   = (MQ / 32) * tiles_n;       // 256 * 8 = 2048
    gemm_bf16_wmma<<<waves / 8, 256, 0, stream>>>(obf, Dc, woutb, Dc, out_b,
                                                  out + Dc, 2 * Dc, MQ, Dc, tiles_n);
  }
}
